// SoftHd_37417755083135
// MI455X (gfx1250) — matrix operand order repaired and checked by reading, not compiled, not run on hardware
//
#include <hip/hip_runtime.h>
#include <stdint.h>

// ---------------------------------------------------------------------------
// Soft Hausdorff distance, fused bf16x3 WMMA implementation for gfx1250.
//   dist[b,i,j] = ||x1[b,i]-x2[b,j]||^2 = sq1[i]+sq2[j]-2*x1[i].x2[j]
//   out[b] = (sum_j min_i dist + sum_i min_j dist) / min(sz1,sz2)
// Row/col mins combined via uint-bit atomicMin (valid: dist >= 0).
// ---------------------------------------------------------------------------

typedef __attribute__((ext_vector_type(16))) __bf16 v16bf;
typedef __attribute__((ext_vector_type(8)))  float  v8f;

#define B_    64
#define N1_   512
#define N2_   512
#define D_    128
#define TM    128            // x1 rows per workgroup
#define CB    256            // dist columns per workgroup
#define NCH   128            // x2 rows (dist columns) per LDS chunk
#define NT1   (N1_ / TM)     // 4 row tiles
#define NTC   (N2_ / CB)     // 2 col blocks
#define AST   136            // LDS row stride in bf16 elems (pad 8 vs 128)
#define BST   136
#define INFB  0x7F800000u    // +inf bit pattern

union Frag { v16bf v; unsigned u[8]; };

// ---------------------------------------------------------------------------
__global__ __launch_bounds__(256)
void softhd_init(unsigned* __restrict__ mins) {
    int idx = blockIdx.x * 256 + threadIdx.x;
    if (idx < B_ * (N1_ + N2_)) mins[idx] = INFB;
}

// ---------------------------------------------------------------------------
__global__ __launch_bounds__(256)
void softhd_tiles(const float* __restrict__ x1, const float* __restrict__ x2,
                  const int* __restrict__ sz1, const int* __restrict__ sz2,
                  unsigned* __restrict__ colmin_g, unsigned* __restrict__ rowmin_g)
{
    __shared__ __bf16   aHi[TM * AST];
    __shared__ __bf16   aLo[TM * AST];
    __shared__ __bf16   bHi[NCH * BST];
    __shared__ __bf16   bLo[NCH * BST];
    __shared__ float    sq1s[TM];
    __shared__ float    sq2s[NCH];
    __shared__ unsigned rowmin[TM];
    __shared__ unsigned colmin[CB];

    const int tid = threadIdx.x;
    const int b   = blockIdx.x >> 3;          // 8 tiles per batch
    const int tm  = (blockIdx.x >> 1) & 3;
    const int tn  = blockIdx.x & 1;
    const int m0  = tm * TM;
    const int c0  = tn * CB;
    const int s1  = sz1[b];
    const int s2  = sz2[b];
    if (m0 >= s1 || c0 >= s2) return;         // uniform: tile fully invalid

    for (int j = tid; j < CB; j += 256) colmin[j] = INFB;
    if (tid < TM) { rowmin[tid] = INFB; sq1s[tid] = 0.0f; }
    __syncthreads();

    // ---- load x1 tile, split f32 -> bf16 hi/lo, accumulate row norms -------
    {
        const float* x1p = x1 + ((size_t)b * N1_ + m0) * D_;
        #pragma unroll
        for (int it = 0; it < (TM * D_ / 4) / 256; ++it) {
            int idx = it * 256 + tid;                 // float4 index
            int row = idx / (D_ / 4);
            int col = (idx % (D_ / 4)) * 4;
            const float4 v = ((const float4*)x1p)[idx];
            float vs[4] = { v.x, v.y, v.z, v.w };
            float ss = 0.0f;
            #pragma unroll
            for (int u = 0; u < 4; ++u) {
                float  f = vs[u];
                __bf16 h = (__bf16)f;
                __bf16 l = (__bf16)(f - (float)h);
                aHi[row * AST + col + u] = h;
                aLo[row * AST + col + u] = l;
                ss += f * f;
            }
            atomicAdd(&sq1s[row], ss);                // ds_add_f32
        }
    }
    __syncthreads();

    const int lane = tid & 31;
    const int wv   = tid >> 5;            // wave 0..7 -> 16-row strip
    const int half = lane >> 4;           // 0 | 1
    const int l16  = lane & 15;
    const int mrow = wv * 16;

    // ---- hoist per-lane row-constant state: A frags, row norms, validity ---
    Frag fAh[4], fAl[4];
    {
        const int arow = (mrow + l16) * AST;
        #pragma unroll
        for (int kc = 0; kc < 4; ++kc) {
            #pragma unroll
            for (int p = 0; p < 8; ++p) {
                // 16-bit A 16x32 layout: K = grp*16 + half*8 + (p%4)*2
                int k = kc * 32 + ((p >> 2) << 4) + (half << 3) + ((p & 3) << 1);
                fAh[kc].u[p] = *(const unsigned*)&aHi[arow + k];
                fAl[kc].u[p] = *(const unsigned*)&aLo[arow + k];
            }
        }
    }
    float q1[8];       // sq1 for this lane's 8 accumulator rows
    bool  vrow[8];     // row validity
    float rmin[8];     // running masked row-min (registers, flushed at end)
    #pragma unroll
    for (int r = 0; r < 8; ++r) {
        const int ml = mrow + r + (half << 3);
        q1[r]   = sq1s[ml];
        vrow[r] = (m0 + ml) < s1;
        rmin[r] = __builtin_inff();
    }

    const int lastchunk = ((s2 < c0 + CB ? s2 : c0 + CB) - c0 + NCH - 1) / NCH;
    for (int nc = 0; nc < lastchunk; ++nc) {
        const int cstart = c0 + nc * NCH;
        __syncthreads();                              // chunk buffers free
        if (tid < NCH) sq2s[tid] = 0.0f;
        __syncthreads();

        // ---- load x2 chunk [NCH x D], split hi/lo, accumulate col norms ----
        const float* x2p = x2 + ((size_t)b * N2_ + cstart) * D_;
        #pragma unroll
        for (int it = 0; it < (NCH * D_ / 4) / 256; ++it) {
            int idx = it * 256 + tid;
            int row = idx / (D_ / 4);
            int col = (idx % (D_ / 4)) * 4;
            const float4 v = ((const float4*)x2p)[idx];
            float vs[4] = { v.x, v.y, v.z, v.w };
            float ss = 0.0f;
            #pragma unroll
            for (int u = 0; u < 4; ++u) {
                float  f = vs[u];
                __bf16 h = (__bf16)f;
                __bf16 l = (__bf16)(f - (float)h);
                bHi[row * BST + col + u] = h;
                bLo[row * BST + col + u] = l;
                ss += f * f;
            }
            atomicAdd(&sq2s[row], ss);
        }
        __syncthreads();

        // ---- 16x32 dist tiles: two independent accumulator chains ----------
        // (24 WMMAs / iter; two chains interleave to hide XDL + LDS latency)
        for (int nt = 0; nt < NCH / 32; ++nt) {
            const int ncolb = nt * 32;
            v8f acc0 = {};
            v8f acc1 = {};
            #pragma unroll
            for (int kc = 0; kc < 4; ++kc) {
                Frag fB0h, fB0l, fB1h, fB1l;
                // 16-bit B 32x16 layout: K = grp*16 + half*8 + (p%4)*2
                const int brow0 = (ncolb + l16) * BST + kc * 32 + (half << 3);
                const int brow1 = brow0 + 16 * BST;
                #pragma unroll
                for (int p = 0; p < 8; ++p) {
                    const int ko = ((p >> 2) << 4) + ((p & 3) << 1);
                    fB0h.u[p] = *(const unsigned*)&bHi[brow0 + ko];
                    fB0l.u[p] = *(const unsigned*)&bLo[brow0 + ko];
                    fB1h.u[p] = *(const unsigned*)&bHi[brow1 + ko];
                    fB1l.u[p] = *(const unsigned*)&bLo[brow1 + ko];
                }
                acc0 = __builtin_amdgcn_wmma_f32_16x16x32_bf16(
                           false, fAh[kc].v, false, fB0h.v, (short)0, acc0, false, false);
                acc1 = __builtin_amdgcn_wmma_f32_16x16x32_bf16(
                           false, fAh[kc].v, false, fB1h.v, (short)0, acc1, false, false);
                acc0 = __builtin_amdgcn_wmma_f32_16x16x32_bf16(
                           false, fAh[kc].v, false, fB0l.v, (short)0, acc0, false, false);
                acc1 = __builtin_amdgcn_wmma_f32_16x16x32_bf16(
                           false, fAh[kc].v, false, fB1l.v, (short)0, acc1, false, false);
                acc0 = __builtin_amdgcn_wmma_f32_16x16x32_bf16(
                           false, fAl[kc].v, false, fB0h.v, (short)0, acc0, false, false);
                acc1 = __builtin_amdgcn_wmma_f32_16x16x32_bf16(
                           false, fAl[kc].v, false, fB1h.v, (short)0, acc1, false, false);
            }

            // ---- dist + masked mins (C/D layout: VGPR r -> M=r+8*half) ------
            {
                const int   jg  = cstart + ncolb + l16;
                const bool  vj  = (jg < s2);
                const float q2  = sq2s[ncolb + l16];
                float cmin = __builtin_inff();
                #pragma unroll
                for (int r = 0; r < 8; ++r) {
                    float d = q1[r] + q2 - 2.0f * acc0[r];
                    d = d > 0.0f ? d : 0.0f;
                    float dr = vj      ? d : __builtin_inff();
                    float dc = vrow[r] ? d : __builtin_inff();
                    rmin[r] = dr < rmin[r] ? dr : rmin[r];
                    cmin    = dc < cmin    ? dc : cmin;
                }
                atomicMin(&colmin[nc * NCH + ncolb + l16], __float_as_uint(cmin));
            }
            {
                const int   jg  = cstart + ncolb + 16 + l16;
                const bool  vj  = (jg < s2);
                const float q2  = sq2s[ncolb + 16 + l16];
                float cmin = __builtin_inff();
                #pragma unroll
                for (int r = 0; r < 8; ++r) {
                    float d = q1[r] + q2 - 2.0f * acc1[r];
                    d = d > 0.0f ? d : 0.0f;
                    float dr = vj      ? d : __builtin_inff();
                    float dc = vrow[r] ? d : __builtin_inff();
                    rmin[r] = dr < rmin[r] ? dr : rmin[r];
                    cmin    = dc < cmin    ? dc : cmin;
                }
                atomicMin(&colmin[nc * NCH + ncolb + 16 + l16], __float_as_uint(cmin));
            }
        }
    }

    // ---- flush register row-mins: lanes -> LDS, then LDS -> global ---------
    #pragma unroll
    for (int r = 0; r < 8; ++r)
        atomicMin(&rowmin[mrow + r + (half << 3)], __float_as_uint(rmin[r]));
    __syncthreads();

    if (tid < TM) {
        const int ig = m0 + tid;
        if (ig < s1) atomicMin(&rowmin_g[(size_t)b * N1_ + ig], rowmin[tid]);
    }
    for (int j = tid; j < CB; j += 256) {
        const int jg = c0 + j;
        if (jg < s2) atomicMin(&colmin_g[(size_t)b * N2_ + jg], colmin[j]);
    }
}

// ---------------------------------------------------------------------------
__global__ __launch_bounds__(256)
void softhd_final(const unsigned* __restrict__ colmin_g,
                  const unsigned* __restrict__ rowmin_g,
                  const int* __restrict__ sz1, const int* __restrict__ sz2,
                  float* __restrict__ out)
{
    __shared__ float red[256];
    const int b  = blockIdx.x;
    const int s1 = sz1[b];
    const int s2 = sz2[b];

    float acc = 0.0f;
    for (int j = threadIdx.x; j < N2_; j += 256)
        if (j < s2) acc += __uint_as_float(colmin_g[(size_t)b * N2_ + j]);
    for (int i = threadIdx.x; i < N1_; i += 256)
        if (i < s1) acc += __uint_as_float(rowmin_g[(size_t)b * N1_ + i]);
    red[threadIdx.x] = acc;
    __syncthreads();
    #pragma unroll
    for (int s = 128; s > 0; s >>= 1) {
        if (threadIdx.x < s) red[threadIdx.x] += red[threadIdx.x + s];
        __syncthreads();
    }
    if (threadIdx.x == 0) {
        float denom = (float)(s1 < s2 ? s1 : s2);
        out[b] = red[0] / denom;
    }
}

// ---------------------------------------------------------------------------
extern "C" void kernel_launch(void* const* d_in, const int* in_sizes, int n_in,
                              void* d_out, int out_size, void* d_ws, size_t ws_size,
                              hipStream_t stream)
{
    (void)in_sizes; (void)n_in; (void)out_size; (void)ws_size;
    const float* x1  = (const float*)d_in[0];
    const float* x2  = (const float*)d_in[1];
    const int*   sz1 = (const int*)d_in[2];
    const int*   sz2 = (const int*)d_in[3];
    float*       out = (float*)d_out;

    unsigned* colmin = (unsigned*)d_ws;                                    // B*N2 u32
    unsigned* rowmin = colmin + (size_t)B_ * N2_;                          // B*N1 u32

    softhd_init <<<(B_ * (N1_ + N2_) + 255) / 256, 256, 0, stream>>>(colmin);
    softhd_tiles<<<B_ * NT1 * NTC,              256, 0, stream>>>(x1, x2, sz1, sz2,
                                                                  colmin, rowmin);
    softhd_final<<<B_,                          256, 0, stream>>>(colmin, rowmin,
                                                                  sz1, sz2, out);
}
